// Attention_78297253806705
// MI455X (gfx1250) — matrix operand order repaired and checked by reading, not compiled, not run on hardware
//
#include <hip/hip_runtime.h>

// ---------------------------------------------------------------------------
// Attention block for MI455X (gfx1250, wave32, WMMA bf16).
//   x:(1,2048,1024) f32 ; wq,wk,wv,wp:(1024,1024) f32 ; out:(1,2048,1024) f32
// Pipeline: f32->bf16 convert -> 3 WMMA GEMMs (Q,K,V f32) -> RoPE+repack bf16
//           -> flash attention (WMMA QK^T + online softmax + WMMA PV)
//           -> WMMA output-projection GEMM (f32 out).
// GEMM staging uses CDNA5 async global->LDS loads (ASYNCcnt).
// ---------------------------------------------------------------------------

typedef unsigned short u16;
typedef __bf16 bf16;
typedef __attribute__((ext_vector_type(8)))  bf16 v8bf;
typedef __attribute__((ext_vector_type(16))) bf16 v16bf;
typedef __attribute__((ext_vector_type(8)))  float v8f;

#define SEQ 2048
#define DIMX 1024
#define NH 16
#define HD 64

__device__ __forceinline__ u16 f32_to_bf16_bits(float f) {
  unsigned u = __builtin_bit_cast(unsigned, f);
  u += 0x7FFFu + ((u >> 16) & 1u);   // round to nearest even
  return (u16)(u >> 16);
}

__device__ __forceinline__ v16bf cat16(v8bf lo, v8bf hi) {
  return __builtin_shufflevector(lo, hi, 0,1,2,3,4,5,6,7,8,9,10,11,12,13,14,15);
}

// CDNA5 async global->LDS copy, 16 bytes per lane. Tracked by ASYNCcnt.
// lds_off = low 32 bits of the generic shared pointer.
__device__ __forceinline__ void async_copy_b128(unsigned lds_off, const void* gaddr) {
  asm volatile("global_load_async_to_lds_b128 %0, %1, off"
               :: "v"(lds_off), "v"(gaddr) : "memory");
}
__device__ __forceinline__ void wait_async0() {
  asm volatile("s_wait_asynccnt 0x0" ::: "memory");
}

// ---------------------------------------------------------------------------
// Elementwise f32 -> bf16 convert
// ---------------------------------------------------------------------------
__global__ __launch_bounds__(256) void cvt_f32_bf16(const float* __restrict__ in,
                                                    u16* __restrict__ out, int n) {
  int i = blockIdx.x * 256 + threadIdx.x;
  if (i < n) out[i] = f32_to_bf16_bits(in[i]);
}

// ---------------------------------------------------------------------------
// C[M,N] (f32) = A[M,K](bf16, row-major) x W[N,K](bf16, row-major)^T
// Block: 256 threads (8 wave32), tile 128(M) x 128(N), K-step 32.
// Wave grid 4(M) x 2(N); each wave computes 32x64 via 2x4 wmma accumulators
// -> 8 v_wmma_f32_16x16x32_bf16 per 12 ds_load_b128 per k-step.
// ---------------------------------------------------------------------------
__global__ __launch_bounds__(256) void gemm_xwt(const u16* __restrict__ A,
                                                const u16* __restrict__ W,
                                                float* __restrict__ C,
                                                int M, int N, int K) {
  __shared__ u16 As[128 * 32];   // [m][k]  8 KB
  __shared__ u16 Bs[128 * 32];   // [n][k]  8 KB
  const int t    = threadIdx.x;
  const int lane = t & 31;
  const int wid  = t >> 5;
  const int wm   = wid & 3;      // 0..3  -> 32-row strip
  const int wn   = wid >> 2;     // 0..1  -> 64-col strip
  const int m0   = blockIdx.y * 128;
  const int n0   = blockIdx.x * 128;
  const int lh   = lane >> 4;    // half-wave select
  const int ll   = lane & 15;

  v8f acc[2][4] = {};

  // Staging: 128 rows x 32 cols per tile; thread t covers row t>>1,
  // 16-element half (t&1), as two b128 async copies.
  const int srow = t >> 1;             // 0..127
  const int scol = (t & 1) * 16;       // 0 or 16
  const unsigned ldsA0 = (unsigned)(uintptr_t)&As[srow * 32 + scol];
  const unsigned ldsA1 = (unsigned)(uintptr_t)&As[srow * 32 + scol + 8];
  const unsigned ldsB0 = (unsigned)(uintptr_t)&Bs[srow * 32 + scol];
  const unsigned ldsB1 = (unsigned)(uintptr_t)&Bs[srow * 32 + scol + 8];
  const u16* gA = A + (m0 + srow) * K + scol;
  const u16* gB = W + (n0 + srow) * K + scol;

  for (int k0 = 0; k0 < K; k0 += 32) {
    __syncthreads();               // previous iteration's reads done
    async_copy_b128(ldsA0, gA);
    async_copy_b128(ldsA1, gA + 8);
    async_copy_b128(ldsB0, gB);
    async_copy_b128(ldsB1, gB + 8);
    gA += 32;
    gB += 32;
    wait_async0();                 // this wave's async writes landed in LDS
    __syncthreads();               // all waves' writes visible

    // A fragments (16x32 bf16): lane ll = row, halves at k = 8*lh and 16+8*lh
    v16bf af[2];
#pragma unroll
    for (int mt = 0; mt < 2; ++mt) {
      const u16* ap = &As[(wm * 32 + mt * 16 + ll) * 32 + lh * 8];
      af[mt] = cat16(*(const v8bf*)ap, *(const v8bf*)(ap + 16));
    }
#pragma unroll
    for (int nt = 0; nt < 4; ++nt) {
      // B fragment (32x16 bf16): lane ll = col n, halves at k = 8*lh and 16+8*lh
      const u16* bp = &Bs[(wn * 64 + nt * 16 + ll) * 32 + lh * 8];
      v16bf bfr = cat16(*(const v8bf*)bp, *(const v8bf*)(bp + 16));
#pragma unroll
      for (int mt = 0; mt < 2; ++mt) {
        acc[mt][nt] = __builtin_amdgcn_wmma_f32_16x16x32_bf16(
            false, af[mt], false, bfr, (short)0, acc[mt][nt], false, false);
      }
    }
  }

#pragma unroll
  for (int mt = 0; mt < 2; ++mt)
#pragma unroll
    for (int nt = 0; nt < 4; ++nt)
#pragma unroll
      for (int r = 0; r < 8; ++r) {
        int row = m0 + wm * 32 + mt * 16 + r + lh * 8;
        int col = n0 + wn * 64 + nt * 16 + ll;
        C[row * N + col] = acc[mt][nt][r];
      }
}

// ---------------------------------------------------------------------------
// RoPE + repack: Qf/Kf/Vf (f32 [n][1024]) ->
//   Qh,Kh bf16 [h][n][64]   (roped)
//   Vt    bf16 [h][64][n]   (transposed for PV B-fragments)
// ---------------------------------------------------------------------------
__global__ __launch_bounds__(256) void rope_pack(const float* __restrict__ Qf,
                                                 const float* __restrict__ Kf,
                                                 const float* __restrict__ Vf,
                                                 u16* __restrict__ Qh,
                                                 u16* __restrict__ Kh,
                                                 u16* __restrict__ Vt) {
  int i = blockIdx.x * 256 + threadIdx.x;   // over SEQ*DIMX
  int n = i >> 10;
  int c = i & 1023;
  int h = c >> 6;
  int d = c & 63;
  int fi = d & 31;
  // inv_freq = 10000^(-2*fi/64) = exp(-ln(10000)*2*fi/64)
  float ang = (float)n * __expf((float)(2 * fi) * (-9.210340371976184f / 64.0f));
  float cs = __cosf(ang), sn = __sinf(ang);
  float q = Qf[i], k = Kf[i];
  float q2, k2;
  if (d < 32) { q2 = q * cs - Qf[i + 32] * sn; k2 = k * cs - Kf[i + 32] * sn; }
  else        { q2 = q * cs + Qf[i - 32] * sn; k2 = k * cs + Kf[i - 32] * sn; }
  Qh[(h * SEQ + n) * HD + d] = f32_to_bf16_bits(q2);
  Kh[(h * SEQ + n) * HD + d] = f32_to_bf16_bits(k2);
  Vt[h * HD * SEQ + d * SEQ + n] = f32_to_bf16_bits(Vf[i]);
}

// ---------------------------------------------------------------------------
// Flash attention: one wave per (head, 16-query block), streaming 32-key
// blocks with online softmax. P is relaid C-layout -> A-layout via a private
// LDS tile (per-wave, DS ops in-order; explicit s_wait_dscnt before reload).
// ---------------------------------------------------------------------------
__global__ __launch_bounds__(128) void flash_attn(const u16* __restrict__ Qh,
                                                  const u16* __restrict__ Kh,
                                                  const u16* __restrict__ Vt,
                                                  u16* __restrict__ Oh) {
  __shared__ u16 Plds[4 * 16 * 32];   // 4 waves x (16x32) bf16 P tile
  const int lane  = threadIdx.x & 31;
  const int wslot = threadIdx.x >> 5;
  const int gw    = blockIdx.x * 4 + wslot;   // 0..2047
  const int head  = gw >> 7;                  // / (SEQ/16)
  const int qb    = gw & 127;
  const int lh    = lane >> 4;
  const int ll    = lane & 15;

  // Q fragments for the two 32-wide k-chunks of hd=64
  const u16* qbase = Qh + (head * SEQ + qb * 16 + ll) * HD;
  v16bf qf[2];
#pragma unroll
  for (int c2 = 0; c2 < 2; ++c2) {
    const u16* p = qbase + c2 * 32 + lh * 8;
    qf[c2] = cat16(*(const v8bf*)p, *(const v8bf*)(p + 16));
  }

  v8f acc[4] = {};
  float m[8], l[8];
#pragma unroll
  for (int r = 0; r < 8; ++r) { m[r] = -3.0e38f; l[r] = 0.0f; }

  u16* pl = &Plds[wslot * 512];
  const u16* kbaseh = Kh + head * SEQ * HD;
  const u16* vbaseh = Vt + head * HD * SEQ;

  for (int kb = 0; kb < SEQ; kb += 32) {
    // S = Q * K^T for 32 keys: two 16x16 tiles, k-dim 64 = 2 chunks
    v8f s[2] = {};
#pragma unroll
    for (int c2 = 0; c2 < 2; ++c2) {
#pragma unroll
      for (int tt = 0; tt < 2; ++tt) {
        const u16* p = kbaseh + (kb + tt * 16 + ll) * HD + c2 * 32 + lh * 8;
        v16bf bfr = cat16(*(const v8bf*)p, *(const v8bf*)(p + 16));
        s[tt] = __builtin_amdgcn_wmma_f32_16x16x32_bf16(
            false, qf[c2], false, bfr, (short)0, s[tt], false, false);
      }
    }

    // Online softmax per row (rows live in 16-lane halves of the C layout)
#pragma unroll
    for (int r = 0; r < 8; ++r) {
      float s0 = s[0][r] * 0.125f;   // 1/sqrt(64)
      float s1 = s[1][r] * 0.125f;
      float v = fmaxf(s0, s1);
      v = fmaxf(v, __shfl_xor(v, 1, 32));
      v = fmaxf(v, __shfl_xor(v, 2, 32));
      v = fmaxf(v, __shfl_xor(v, 4, 32));
      v = fmaxf(v, __shfl_xor(v, 8, 32));
      float mn = fmaxf(m[r], v);
      float corr = __expf(m[r] - mn);
      m[r] = mn;
      l[r] *= corr;
#pragma unroll
      for (int nt = 0; nt < 4; ++nt) acc[nt][r] *= corr;
      float p0 = __expf(s0 - mn);
      float p1 = __expf(s1 - mn);
      float ts = p0 + p1;
      ts += __shfl_xor(ts, 1, 32);
      ts += __shfl_xor(ts, 2, 32);
      ts += __shfl_xor(ts, 4, 32);
      ts += __shfl_xor(ts, 8, 32);
      l[r] += ts;
      int prow = r + lh * 8;
      pl[prow * 32 + ll]      = f32_to_bf16_bits(p0);
      pl[prow * 32 + 16 + ll] = f32_to_bf16_bits(p1);
    }

    // DS ops from one wave are in-order; wait + clobber stops reordering.
    asm volatile("s_wait_dscnt 0x0" ::: "memory");

    // Reload P as an A fragment (16x32 bf16)
    const u16* pp = pl + ll * 32 + lh * 8;
    v16bf pf = cat16(*(const v8bf*)pp, *(const v8bf*)(pp + 16));

    // O += P * V ; V^T layout makes B fragments contiguous along keys
#pragma unroll
    for (int nt = 0; nt < 4; ++nt) {
      const u16* vp = vbaseh + (nt * 16 + ll) * SEQ + kb + lh * 8;
      v16bf vf = cat16(*(const v8bf*)vp, *(const v8bf*)(vp + 16));
      acc[nt] = __builtin_amdgcn_wmma_f32_16x16x32_bf16(
          false, pf, false, vf, (short)0, acc[nt], false, false);
    }
  }

  // Normalize and pack to Oh [n][1024] bf16 (col = head*64 + d)
#pragma unroll
  for (int r = 0; r < 8; ++r) {
    float inv = 1.0f / l[r];
    int row = qb * 16 + r + lh * 8;
#pragma unroll
    for (int nt = 0; nt < 4; ++nt) {
      Oh[row * DIMX + head * HD + nt * 16 + ll] = f32_to_bf16_bits(acc[nt][r] * inv);
    }
  }
}

// ---------------------------------------------------------------------------
// Host-side orchestration
// ---------------------------------------------------------------------------
extern "C" void kernel_launch(void* const* d_in, const int* in_sizes, int n_in,
                              void* d_out, int out_size, void* d_ws, size_t ws_size,
                              hipStream_t stream) {
  (void)in_sizes; (void)n_in; (void)out_size; (void)ws_size;
  const float* x  = (const float*)d_in[0];
  const float* wq = (const float*)d_in[1];
  const float* wk = (const float*)d_in[2];
  const float* wv = (const float*)d_in[3];
  const float* wp = (const float*)d_in[4];

  char* ws = (char*)d_ws;
  const size_t MB = 1024 * 1024;
  u16*   xh  = (u16*)  (ws + 0);        // 4 MiB  x bf16
  u16*   wqh = (u16*)  (ws + 4  * MB);  // 2 MiB
  u16*   wkh = (u16*)  (ws + 6  * MB);  // 2 MiB
  u16*   wvh = (u16*)  (ws + 8  * MB);  // 2 MiB
  u16*   wph = (u16*)  (ws + 10 * MB);  // 2 MiB
  float* Qf  = (float*)(ws + 12 * MB);  // 8 MiB
  float* Kf  = (float*)(ws + 20 * MB);  // 8 MiB
  float* Vf  = (float*)(ws + 28 * MB);  // 8 MiB
  u16*   Qh  = (u16*)  (ws + 36 * MB);  // 4 MiB  [h][n][64]
  u16*   Kh  = (u16*)  (ws + 40 * MB);  // 4 MiB  [h][n][64]
  u16*   Vt  = (u16*)  (ws + 44 * MB);  // 4 MiB  [h][64][n]
  u16*   Oh  = (u16*)  (ws + 48 * MB);  // 4 MiB  [n][1024]

  cvt_f32_bf16<<<(SEQ * DIMX) / 256, 256, 0, stream>>>(x,  xh,  SEQ * DIMX);
  cvt_f32_bf16<<<(DIMX * DIMX) / 256, 256, 0, stream>>>(wq, wqh, DIMX * DIMX);
  cvt_f32_bf16<<<(DIMX * DIMX) / 256, 256, 0, stream>>>(wk, wkh, DIMX * DIMX);
  cvt_f32_bf16<<<(DIMX * DIMX) / 256, 256, 0, stream>>>(wv, wvh, DIMX * DIMX);
  cvt_f32_bf16<<<(DIMX * DIMX) / 256, 256, 0, stream>>>(wp, wph, DIMX * DIMX);

  dim3 gg(DIMX / 128, SEQ / 128);
  gemm_xwt<<<gg, 256, 0, stream>>>(xh, wqh, Qf, SEQ, DIMX, DIMX);
  gemm_xwt<<<gg, 256, 0, stream>>>(xh, wkh, Kf, SEQ, DIMX, DIMX);
  gemm_xwt<<<gg, 256, 0, stream>>>(xh, wvh, Vf, SEQ, DIMX, DIMX);

  rope_pack<<<(SEQ * DIMX) / 256, 256, 0, stream>>>(Qf, Kf, Vf, Qh, Kh, Vt);

  flash_attn<<<(NH * (SEQ / 16)) / 4, 128, 0, stream>>>(Qh, Kh, Vt, Oh);

  gemm_xwt<<<gg, 256, 0, stream>>>(Oh, wph, (float*)d_out, SEQ, DIMX, DIMX);
}
